// GATWeightedClassifier_33260226740829
// MI455X (gfx1250) — matrix operand order repaired and checked by reading, not compiled, not run on hardware
//
#include <hip/hip_runtime.h>
#include <math.h>

#define N_NODES  100000            // divisible by 32 -> no tile guards needed
#define N_EDGES  1600000
#define TOT_E    (N_EDGES + N_NODES)   // edges + self loops
#define IN_C     128
#define HID      32
#define HEADS    4
#define OUT_C    16
#define G_GRAPHS 128
#define NEG_SLOPE 0.2f

typedef __attribute__((ext_vector_type(16))) _Float16 v16h;
typedef __attribute__((ext_vector_type(8)))  _Float16 v8h;
typedef __attribute__((ext_vector_type(8)))  float    v8f;

// ---------------- small helpers ----------------
__device__ __forceinline__ float lrelu(float v) { return v > 0.f ? v : NEG_SLOPE * v; }
__device__ __forceinline__ float elu1(float v)  { return v > 0.f ? v : (__expf(v) - 1.f); }

__device__ __forceinline__ void atomic_max_f32(float* addr, float val) {
  // IEEE-monotonic int-bits trick (values may be negative)
  if (val >= 0.0f)
    atomicMax((int*)addr, __float_as_int(val));
  else
    atomicMin((unsigned int*)addr, __float_as_uint(val));
}

__device__ __forceinline__ void edge_sd(const long long* __restrict__ ei, int e, int& s, int& d) {
  if (e < N_EDGES) { s = (int)ei[e]; d = (int)ei[N_EDGES + e]; }
  else             { s = e - N_EDGES; d = s; }
}

__global__ void fill_kernel(float* __restrict__ p, float v, int n) {
  int i = blockIdx.x * blockDim.x + threadIdx.x;
  if (i < n) p[i] = v;
}

// ---------------- weight pre-swizzle into WMMA B-fragment order ----------------
// B fragment (16x16x32 f16): lane L holds col n = ct*16 + (L&15); elems e=0..15 are
// K = kt*32 + (e>>3)*16 + (L>>4)*8 + (e&7). Packed contiguously so GEMM does one aligned v16h load.
__global__ void pack_weights_kernel(const float* __restrict__ W1, const float* __restrict__ W2,
                                    _Float16* __restrict__ Bp1, _Float16* __restrict__ Bp2) {
  int t = threadIdx.x;
  for (int idx = t; idx < 8 * 4 * 32 * 16; idx += 256) {   // W1: 8 col-tiles x 4 k-tiles
    int e = idx & 15, lane = (idx >> 4) & 31, kt = (idx >> 9) & 3, ct = idx >> 11;
    int k = kt * 32 + ((e >> 3) << 4) + ((lane >> 4) << 3) + (e & 7);
    int n = ct * 16 + (lane & 15);
    Bp1[idx] = (_Float16)W1[k * (HEADS * HID) + n];
  }
  for (int idx = t; idx < 2 * 4 * 32 * 16; idx += 256) {   // W2: 2 col-tiles x 4 k-tiles
    int e = idx & 15, lane = (idx >> 4) & 31, kt = (idx >> 9) & 3, ct = idx >> 11;
    int k = kt * 32 + ((e >> 3) << 4) + ((lane >> 4) << 3) + (e & 7);
    int n = ct * 16 + (lane & 15);
    Bp2[idx] = (_Float16)W2[k * HID + n];
  }
}

// ---------------- GEMM1: h1[N,128] = (x * nw) @ W1, WMMA f16->f32 ----------------
// block = 256 threads = 8 waves; block owns 32 rows; wave w owns 16-col tile w and
// both 16-row tiles (one B-fragment load feeds two WMMAs).
__global__ void __launch_bounds__(256) gemm1_kernel(const float* __restrict__ x,
                                                    const float* __restrict__ nw,
                                                    const _Float16* __restrict__ Bp1,
                                                    float* __restrict__ h1) {
  __shared__ __align__(16) _Float16 As[32][IN_C];
  const int row0 = blockIdx.x * 32;          // grid = 3125, 3125*32 == N exactly
  const int tid  = threadIdx.x;
  { // stage A tile (apply node_weights, convert to f16); 16 elems per thread
    int r  = tid >> 3;
    int c0 = (tid & 7) * 16;
    int n  = row0 + r;
    float w = nw[n];
    const float4* p = (const float4*)(x + (size_t)n * IN_C + c0);
    #pragma unroll
    for (int q = 0; q < 4; ++q) {
      float4 v = p[q];
      As[r][c0 + q * 4 + 0] = (_Float16)(v.x * w);
      As[r][c0 + q * 4 + 1] = (_Float16)(v.y * w);
      As[r][c0 + q * 4 + 2] = (_Float16)(v.z * w);
      As[r][c0 + q * 4 + 3] = (_Float16)(v.w * w);
    }
  }
  __syncthreads();
  const int wave = tid >> 5;
  const int lane = tid & 31;
  const int m    = lane & 15;
  const int koff = (lane >> 4) * 8;
  v8f acc0 = {}, acc1 = {};
  #pragma unroll
  for (int kt = 0; kt < 4; ++kt) {
    v16h b = *(const v16h*)(Bp1 + ((((wave * 4 + kt) * 32) + lane) << 4));
    // A fragment: per ISA, lane holds K=[koff..koff+7] and K=[16+koff..16+koff+7]
    v8h lo0 = *(const v8h*)&As[m][kt * 32 + koff];
    v8h hi0 = *(const v8h*)&As[m][kt * 32 + 16 + koff];
    v8h lo1 = *(const v8h*)&As[16 + m][kt * 32 + koff];
    v8h hi1 = *(const v8h*)&As[16 + m][kt * 32 + 16 + koff];
    v16h a0, a1;
    #pragma unroll
    for (int i = 0; i < 8; ++i) { a0[i] = lo0[i]; a0[i + 8] = hi0[i];
                                  a1[i] = lo1[i]; a1[i + 8] = hi1[i]; }
    acc0 = __builtin_amdgcn_wmma_f32_16x16x32_f16(false, a0, false, b, (short)0, acc0, false, false);
    acc1 = __builtin_amdgcn_wmma_f32_16x16x32_f16(false, a1, false, b, (short)0, acc1, false, false);
  }
  const int col   = wave * 16 + m;
  const int rbase = (lane >> 4) * 8;
  #pragma unroll
  for (int r = 0; r < 8; ++r) {
    h1[(size_t)(row0 + rbase + r) * IN_C + col]      = acc0[r];
    h1[(size_t)(row0 + 16 + rbase + r) * IN_C + col] = acc1[r];
  }
}

// ---------------- GEMM2: h2[N,32] = out1[N,128] @ W2, WMMA ----------------
// block = 64 threads = 2 waves; block owns 32 rows; wave w owns col tile w.
__global__ void __launch_bounds__(64) gemm2_kernel(const float* __restrict__ A,
                                                   const _Float16* __restrict__ Bp2,
                                                   float* __restrict__ h2) {
  __shared__ __align__(16) _Float16 As[32][IN_C];
  const int row0 = blockIdx.x * 32;
  const int tid  = threadIdx.x;
  { // stage 32x128 f32 -> f16; 64 elems per thread
    int r  = tid >> 1;
    int c0 = (tid & 1) * 64;
    int n  = row0 + r;
    const float4* p = (const float4*)(A + (size_t)n * IN_C + c0);
    #pragma unroll
    for (int q = 0; q < 16; ++q) {
      float4 v = p[q];
      As[r][c0 + q * 4 + 0] = (_Float16)v.x;
      As[r][c0 + q * 4 + 1] = (_Float16)v.y;
      As[r][c0 + q * 4 + 2] = (_Float16)v.z;
      As[r][c0 + q * 4 + 3] = (_Float16)v.w;
    }
  }
  __syncthreads();
  const int wave = tid >> 5;     // 0,1 -> two 16-col tiles of HID=32
  const int lane = tid & 31;
  const int m    = lane & 15;
  const int koff = (lane >> 4) * 8;
  v8f acc0 = {}, acc1 = {};
  #pragma unroll
  for (int kt = 0; kt < 4; ++kt) {
    v16h b = *(const v16h*)(Bp2 + ((((wave * 4 + kt) * 32) + lane) << 4));
    v8h lo0 = *(const v8h*)&As[m][kt * 32 + koff];
    v8h hi0 = *(const v8h*)&As[m][kt * 32 + 16 + koff];
    v8h lo1 = *(const v8h*)&As[16 + m][kt * 32 + koff];
    v8h hi1 = *(const v8h*)&As[16 + m][kt * 32 + 16 + koff];
    v16h a0, a1;
    #pragma unroll
    for (int i = 0; i < 8; ++i) { a0[i] = lo0[i]; a0[i + 8] = hi0[i];
                                  a1[i] = lo1[i]; a1[i + 8] = hi1[i]; }
    acc0 = __builtin_amdgcn_wmma_f32_16x16x32_f16(false, a0, false, b, (short)0, acc0, false, false);
    acc1 = __builtin_amdgcn_wmma_f32_16x16x32_f16(false, a1, false, b, (short)0, acc1, false, false);
  }
  const int col   = wave * 16 + m;
  const int rbase = (lane >> 4) * 8;
  #pragma unroll
  for (int r = 0; r < 8; ++r) {
    h2[(size_t)(row0 + rbase + r) * HID + col]      = acc0[r];
    h2[(size_t)(row0 + 16 + rbase + r) * HID + col] = acc1[r];
  }
}

// ---------------- attention projections ----------------
__global__ void att1_kernel(const float* __restrict__ h1, const float* __restrict__ asr,
                            const float* __restrict__ ads, float* __restrict__ a_src,
                            float* __restrict__ a_dst) {
  int n = blockIdx.x * blockDim.x + threadIdx.x;
  if (n >= N_NODES) return;
  const float* row = h1 + (size_t)n * IN_C;
  #pragma unroll
  for (int h = 0; h < HEADS; ++h) {
    float ss = 0.f, dd = 0.f;
    #pragma unroll
    for (int c = 0; c < HID; ++c) {
      float v = row[h * HID + c];
      ss += v * asr[h * HID + c];
      dd += v * ads[h * HID + c];
    }
    a_src[n * HEADS + h] = ss;
    a_dst[n * HEADS + h] = dd;
  }
}

__global__ void att2_kernel(const float* __restrict__ h2, const float* __restrict__ asr,
                            const float* __restrict__ ads, float* __restrict__ a_src,
                            float* __restrict__ a_dst) {
  int n = blockIdx.x * blockDim.x + threadIdx.x;
  if (n >= N_NODES) return;
  const float* row = h2 + (size_t)n * HID;
  float ss = 0.f, dd = 0.f;
  #pragma unroll
  for (int c = 0; c < HID; ++c) { ss += row[c] * asr[c]; dd += row[c] * ads[c]; }
  a_src[n] = ss;
  a_dst[n] = dd;
}

// ---------------- layer-1 edge passes (segment softmax + aggregate) ----------------
__global__ void edge_max1_kernel(const long long* __restrict__ ei, const float* __restrict__ a_src,
                                 const float* __restrict__ a_dst, float* __restrict__ emax) {
  int e = blockIdx.x * blockDim.x + threadIdx.x;
  if (e >= TOT_E) return;
  int s, d; edge_sd(ei, e, s, d);
  #pragma unroll
  for (int h = 0; h < HEADS; ++h) {
    float v = lrelu(a_src[s * HEADS + h] + a_dst[d * HEADS + h]);
    atomic_max_f32(&emax[d * HEADS + h], v);
  }
}

__global__ void edge_sum1_kernel(const long long* __restrict__ ei, const float* __restrict__ a_src,
                                 const float* __restrict__ a_dst, const float* __restrict__ emax,
                                 float* __restrict__ esum) {
  int e = blockIdx.x * blockDim.x + threadIdx.x;
  if (e >= TOT_E) return;
  int s, d; edge_sd(ei, e, s, d);
  #pragma unroll
  for (int h = 0; h < HEADS; ++h) {
    float v = lrelu(a_src[s * HEADS + h] + a_dst[d * HEADS + h]);
    atomicAdd(&esum[d * HEADS + h], __expf(v - emax[d * HEADS + h]));
  }
}

__global__ void edge_aggr1_kernel(const long long* __restrict__ ei, const float* __restrict__ a_src,
                                  const float* __restrict__ a_dst, const float* __restrict__ emax,
                                  const float* __restrict__ esum, const float* __restrict__ h1,
                                  float* __restrict__ out1) {
  int idx = blockIdx.x * blockDim.x + threadIdx.x;   // (edge, head) pairs
  if (idx >= TOT_E * HEADS) return;
  int e = idx >> 2, h = idx & 3;
  int s, d; edge_sd(ei, e, s, d);
  float v = lrelu(a_src[s * HEADS + h] + a_dst[d * HEADS + h]);
  float alpha = __expf(v - emax[d * HEADS + h]) / esum[d * HEADS + h];
  const float4* hp = (const float4*)(h1 + (size_t)s * IN_C + h * HID);
  __builtin_prefetch(hp, 0, 0);
  float* op = out1 + (size_t)d * IN_C + h * HID;
  #pragma unroll
  for (int q = 0; q < 8; ++q) {
    float4 hv = hp[q];
    atomicAdd(op + q * 4 + 0, hv.x * alpha);
    atomicAdd(op + q * 4 + 1, hv.y * alpha);
    atomicAdd(op + q * 4 + 2, hv.z * alpha);
    atomicAdd(op + q * 4 + 3, hv.w * alpha);
  }
}

__global__ void post1_kernel(float* __restrict__ out1, const float* __restrict__ b1) {
  int i = blockIdx.x * blockDim.x + threadIdx.x;
  if (i >= N_NODES * IN_C) return;
  out1[i] = elu1(out1[i] + b1[i & (IN_C - 1)]);
}

// ---------------- layer-2 edge passes (1 head) ----------------
__global__ void edge_max2_kernel(const long long* __restrict__ ei, const float* __restrict__ a_src,
                                 const float* __restrict__ a_dst, float* __restrict__ emax) {
  int e = blockIdx.x * blockDim.x + threadIdx.x;
  if (e >= TOT_E) return;
  int s, d; edge_sd(ei, e, s, d);
  atomic_max_f32(&emax[d], lrelu(a_src[s] + a_dst[d]));
}

__global__ void edge_sum2_kernel(const long long* __restrict__ ei, const float* __restrict__ a_src,
                                 const float* __restrict__ a_dst, const float* __restrict__ emax,
                                 float* __restrict__ esum) {
  int e = blockIdx.x * blockDim.x + threadIdx.x;
  if (e >= TOT_E) return;
  int s, d; edge_sd(ei, e, s, d);
  float v = lrelu(a_src[s] + a_dst[d]);
  atomicAdd(&esum[d], __expf(v - emax[d]));
}

__global__ void edge_aggr2_kernel(const long long* __restrict__ ei, const float* __restrict__ a_src,
                                  const float* __restrict__ a_dst, const float* __restrict__ emax,
                                  const float* __restrict__ esum, const float* __restrict__ h2,
                                  float* __restrict__ out2) {
  int e = blockIdx.x * blockDim.x + threadIdx.x;
  if (e >= TOT_E) return;
  int s, d; edge_sd(ei, e, s, d);
  float v = lrelu(a_src[s] + a_dst[d]);
  float alpha = __expf(v - emax[d]) / esum[d];
  const float4* hp = (const float4*)(h2 + (size_t)s * HID);
  __builtin_prefetch(hp, 0, 0);
  float* op = out2 + (size_t)d * HID;
  #pragma unroll
  for (int q = 0; q < 8; ++q) {
    float4 hv = hp[q];
    atomicAdd(op + q * 4 + 0, hv.x * alpha);
    atomicAdd(op + q * 4 + 1, hv.y * alpha);
    atomicAdd(op + q * 4 + 2, hv.z * alpha);
    atomicAdd(op + q * 4 + 3, hv.w * alpha);
  }
}

// fused: out2 -> +b2 -> elu -> global mean-pool accumulation
__global__ void post2_pool_kernel(const float* __restrict__ out2, const long long* __restrict__ batch,
                                  const float* __restrict__ b2, float* __restrict__ pooled,
                                  float* __restrict__ cnt) {
  int n = blockIdx.x * blockDim.x + threadIdx.x;
  if (n >= N_NODES) return;
  int b = (int)batch[n];
  #pragma unroll
  for (int c = 0; c < HID; ++c) {
    float v = elu1(out2[(size_t)n * HID + c] + b2[c]);
    atomicAdd(&pooled[b * HID + c], v);
  }
  atomicAdd(&cnt[b], 1.0f);
}

__global__ void final_kernel(const float* __restrict__ pooled, const float* __restrict__ cnt,
                             const float* __restrict__ fcW, const float* __restrict__ fcb,
                             float* __restrict__ out) {
  int t = blockIdx.x * blockDim.x + threadIdx.x;
  if (t >= G_GRAPHS * OUT_C) return;
  int g = t >> 4, oc = t & 15;
  float inv = 1.f / fmaxf(cnt[g], 1.f);
  float acc = fcb[oc];
  #pragma unroll
  for (int c = 0; c < HID; ++c)
    acc += (pooled[g * HID + c] * inv) * fcW[c * OUT_C + oc];
  out[t] = acc;
}

// ---------------- host orchestration ----------------
extern "C" void kernel_launch(void* const* d_in, const int* in_sizes, int n_in,
                              void* d_out, int out_size, void* d_ws, size_t ws_size,
                              hipStream_t stream) {
  (void)in_sizes; (void)n_in; (void)out_size; (void)ws_size;
  const float*     x     = (const float*)d_in[0];
  const long long* ei    = (const long long*)d_in[1];
  const long long* batch = (const long long*)d_in[2];
  const float*     nw    = (const float*)d_in[3];
  const float*     W1    = (const float*)d_in[4];
  const float*     as1   = (const float*)d_in[5];
  const float*     ad1   = (const float*)d_in[6];
  const float*     b1    = (const float*)d_in[7];
  const float*     W2    = (const float*)d_in[8];
  const float*     as2   = (const float*)d_in[9];
  const float*     ad2   = (const float*)d_in[10];
  const float*     b2    = (const float*)d_in[11];
  const float*     fcW   = (const float*)d_in[12];
  const float*     fcb   = (const float*)d_in[13];
  float* out = (float*)d_out;

  // workspace layout (floats). Region A is reused between layers.
  float*    ws      = (float*)d_ws;
  const size_t Nf   = N_NODES;
  float*    regA    = ws;                       // 128*N  : h1, later layer-2 buffers
  float*    out1    = ws + 128 * Nf;            // 128*N
  float*    a_src1  = ws + 256 * Nf;            // 4*N
  float*    a_dst1  = a_src1 + 4 * Nf;          // 4*N
  float*    emax1   = a_dst1 + 4 * Nf;          // 4*N
  float*    esum1   = emax1 + 4 * Nf;           // 4*N
  float*    pooled  = esum1 + 4 * Nf;           // 128*32
  float*    cnt     = pooled + G_GRAPHS * HID;  // 128
  _Float16* Bp1     = (_Float16*)(cnt + 128);   // 16384 f16 (32B-aligned offset)
  _Float16* Bp2     = Bp1 + 16384;              // 4096 f16
  // region A aliases (valid after layer-1 edge aggregation finished)
  float* h1     = regA;
  float* h2     = regA;
  float* out2   = regA + 32 * Nf;
  float* a_src2 = regA + 64 * Nf;
  float* a_dst2 = regA + 65 * Nf;
  float* emax2  = regA + 66 * Nf;
  float* esum2  = regA + 67 * Nf;

  const int TB = 256;
  const int rowTiles32 = N_NODES / 32;                    // 3125, exact
  auto blocks = [](int n, int tb) { return (n + tb - 1) / tb; };

  pack_weights_kernel<<<1, 256, 0, stream>>>(W1, W2, Bp1, Bp2);

  fill_kernel<<<blocks(4 * N_NODES, TB), TB, 0, stream>>>(emax1, -INFINITY, 4 * N_NODES);
  fill_kernel<<<blocks(4 * N_NODES, TB), TB, 0, stream>>>(esum1, 0.f, 4 * N_NODES);
  fill_kernel<<<blocks(IN_C * N_NODES, TB), TB, 0, stream>>>(out1, 0.f, IN_C * N_NODES);

  gemm1_kernel<<<rowTiles32, 256, 0, stream>>>(x, nw, Bp1, h1);
  att1_kernel<<<blocks(N_NODES, TB), TB, 0, stream>>>(h1, as1, ad1, a_src1, a_dst1);
  edge_max1_kernel<<<blocks(TOT_E, TB), TB, 0, stream>>>(ei, a_src1, a_dst1, emax1);
  edge_sum1_kernel<<<blocks(TOT_E, TB), TB, 0, stream>>>(ei, a_src1, a_dst1, emax1, esum1);
  edge_aggr1_kernel<<<blocks(TOT_E * HEADS, TB), TB, 0, stream>>>(ei, a_src1, a_dst1, emax1, esum1, h1, out1);
  post1_kernel<<<blocks(IN_C * N_NODES, TB), TB, 0, stream>>>(out1, b1);

  // layer-2 buffers live in region A (h1 dead now)
  fill_kernel<<<blocks(N_NODES, TB), TB, 0, stream>>>(emax2, -INFINITY, N_NODES);
  fill_kernel<<<blocks(N_NODES, TB), TB, 0, stream>>>(esum2, 0.f, N_NODES);
  fill_kernel<<<blocks(HID * N_NODES, TB), TB, 0, stream>>>(out2, 0.f, HID * N_NODES);

  gemm2_kernel<<<rowTiles32, 64, 0, stream>>>(out1, Bp2, h2);
  att2_kernel<<<blocks(N_NODES, TB), TB, 0, stream>>>(h2, as2, ad2, a_src2, a_dst2);
  edge_max2_kernel<<<blocks(TOT_E, TB), TB, 0, stream>>>(ei, a_src2, a_dst2, emax2);
  edge_sum2_kernel<<<blocks(TOT_E, TB), TB, 0, stream>>>(ei, a_src2, a_dst2, emax2, esum2);
  edge_aggr2_kernel<<<blocks(TOT_E, TB), TB, 0, stream>>>(ei, a_src2, a_dst2, emax2, esum2, h2, out2);

  fill_kernel<<<blocks(G_GRAPHS * HID, TB), TB, 0, stream>>>(pooled, 0.f, G_GRAPHS * HID);
  fill_kernel<<<1, 128, 0, stream>>>(cnt, 0.f, G_GRAPHS);

  post2_pool_kernel<<<blocks(N_NODES, TB), TB, 0, stream>>>(out2, batch, b2, pooled, cnt);
  final_kernel<<<blocks(G_GRAPHS * OUT_C, TB), TB, 0, stream>>>(pooled, cnt, fcW, fcb, out);
}
